// HGCNLayer_54906861912490
// MI455X (gfx1250) — matrix operand order repaired and checked by reading, not compiled, not run on hardware
//
#include <hip/hip_runtime.h>

typedef __attribute__((ext_vector_type(16))) _Float16 v16h;
typedef __attribute__((ext_vector_type(8)))  _Float16 v8h;
typedef __attribute__((ext_vector_type(4)))  _Float16 v4h;
typedef __attribute__((ext_vector_type(8)))  float    v8f;
typedef __attribute__((ext_vector_type(4)))  unsigned u32x4;
typedef __attribute__((ext_vector_type(8)))  unsigned u32x8;

#define D 128
#define MIN_NORM 1e-15f
#define BALL_EPS 4e-3f

// wave32 XOR-butterfly sum via ds_swizzle (group-of-32 mode: offset = xor<<10 | and=0x1f)
__device__ __forceinline__ float wave_sum32(float v) {
  v += __int_as_float(__builtin_amdgcn_ds_swizzle(__float_as_int(v), 0x401f)); // xor 16
  v += __int_as_float(__builtin_amdgcn_ds_swizzle(__float_as_int(v), 0x201f)); // xor 8
  v += __int_as_float(__builtin_amdgcn_ds_swizzle(__float_as_int(v), 0x101f)); // xor 4
  v += __int_as_float(__builtin_amdgcn_ds_swizzle(__float_as_int(v), 0x081f)); // xor 2
  v += __int_as_float(__builtin_amdgcn_ds_swizzle(__float_as_int(v), 0x041f)); // xor 1
  return v;
}

__device__ __forceinline__ float artanh_c(float x) {
  x = fminf(fmaxf(x, -1.0f + 1e-7f), 1.0f - 1e-7f);
  return 0.5f * logf((1.0f + x) / (1.0f - x));
}

// ---- kernel 0a: convert x (f32) -> f16 for WMMA A fragments ----
__global__ void __launch_bounds__(256)
cvt_x_f16(const float* __restrict__ x, _Float16* __restrict__ x16, int total4) {
  int i = blockIdx.x * blockDim.x + threadIdx.x;
  if (i >= total4) return;
  const float4 v = ((const float4*)x)[i];
  v4h h; h[0] = (_Float16)v.x; h[1] = (_Float16)v.y; h[2] = (_Float16)v.z; h[3] = (_Float16)v.w;
  ((v4h*)x16)[i] = h;
}

// ---- kernel 0b: W -> f16, and pb = expmap0(b) ----
__global__ void __launch_bounds__(128)
prep_wb(const float* __restrict__ W, const float* __restrict__ b,
        _Float16* __restrict__ W16, float* __restrict__ pb) {
  const int t = threadIdx.x;  // 128 threads
  for (int i = t; i < D * D; i += 128) W16[i] = (_Float16)W[i];
  __shared__ float red[4];
  const float bv = b[t];
  const float s = wave_sum32(bv * bv);
  if ((t & 31) == 0) red[t >> 5] = s;
  __syncthreads();
  const float bn = fmaxf(sqrtf(red[0] + red[1] + red[2] + red[3]), MIN_NORM);
  pb[t] = tanhf(bn) / bn * bv;   // SQRT_C == 1
}

// ---- kernel 1: TDM-staged WMMA GEMM (mx = x @ W^T) fused with mobius_matvec
//      scaling, mobius_add(., expmap0(b)) and logmap0; writes h_t and agg-init.
//      32-row blocks: each wave runs two independent WMMA accumulator chains. ----
__global__ void __launch_bounds__(256)
gemm_hyp_kernel(const float* __restrict__ x, const _Float16* __restrict__ x16,
                const _Float16* __restrict__ W16, const float* __restrict__ pb,
                float* __restrict__ h_t, float* __restrict__ agg, int N) {
  // a_tile: 32 rows x 128 f16, row stride 136 halves (272B) -> conflict-free,
  // produced directly by TDM pad_enable (interval 64 DW = 256B, amount 4 DW = 16B)
  __shared__ __attribute__((aligned(16))) _Float16 a_tile[32 * 136];
  __shared__ float mx_tile[32 * 132];
  const int t = threadIdx.x;
  const int lane = t & 31;
  const int w = t >> 5;              // wave id == output-column tile id
  const int row0 = blockIdx.x * 32;

  // ---- Tensor Data Mover: one descriptor DMAs the whole 32x128 f16 A tile ----
  if (w == 0) {
    const unsigned lds_base = (unsigned)(unsigned long long)&a_tile[0];
    const unsigned long long ga = (unsigned long long)(x16 + (size_t)row0 * D);
    const unsigned nrem = (unsigned)(N - row0);       // OOB rows zero-filled by TDM
    u32x4 g0;
    g0[0] = 1u;                                       // count=1 (valid user D#)
    g0[1] = lds_base;                                 // lds_addr
    g0[2] = (unsigned)(ga & 0xffffffffull);           // global_addr[31:0]
    g0[3] = (unsigned)(ga >> 32) | (2u << 30);        // global_addr[56:32] | type=2
    u32x8 g1;
    g1[0] = (1u << 16)            // data_size = 1 -> 2 bytes
          | (1u << 20)            // pad_enable
          | (5u << 22)            // pad_interval: 64 DWORDs (one 256B row)
          | (3u << 25);           // pad_amount: 4 DWORDs (16B -> 272B LDS stride)
    g1[1] = (128u << 16);                             // tensor_dim0 = 128 (bits 79:48)
    g1[2] = (nrem & 0xffffu) << 16;                   // tensor_dim1 lo (bits 111:80)
    g1[3] = (nrem >> 16) | (128u << 16);              // tensor_dim1 hi | tile_dim0=128
    g1[4] = 32u;                                      // tile_dim1=32, tile_dim2=0 (2D)
    g1[5] = 128u;                                     // tensor_dim0_stride = 128 elems
    g1[6] = 0u;                                       // stride hi / dim1_stride unused
    g1[7] = 0u;
    asm volatile("tensor_load_to_lds %0, %1" :: "s"(g0), "s"(g1) : "memory");
    __builtin_amdgcn_s_wait_tensorcnt(0);
  }
  __syncthreads();

  // ---- WMMA: wave w covers out-cols [16w,16w+16); two M-tiles interleaved ----
  v8f acc0 = {}, acc1 = {};
  const int m  = lane & 15;
  const int ka = (lane >> 4) * 16;   // A: K in {ka..ka+15}
  const int kb = (lane >> 4) * 16;   // B: K in {kb..kb+15}, contiguous in W row
  const int gcol = w * 16 + m;
#pragma unroll
  for (int ks = 0; ks < 4; ++ks) {
    const int K0 = ks * 32;
    const v16h B = *(const v16h*)(W16 + (size_t)gcol * D + K0 + kb); // B = W^T tile
    const v8h a0lo = *(const v8h*)&a_tile[m * 136 + K0 + ka];
    const v8h a0hi = *(const v8h*)&a_tile[m * 136 + K0 + 8 + ka];
    const v8h a1lo = *(const v8h*)&a_tile[(m + 16) * 136 + K0 + ka];
    const v8h a1hi = *(const v8h*)&a_tile[(m + 16) * 136 + K0 + 8 + ka];
    const v16h A0 = __builtin_shufflevector(a0lo, a0hi, 0,1,2,3,4,5,6,7,8,9,10,11,12,13,14,15);
    const v16h A1 = __builtin_shufflevector(a1lo, a1hi, 0,1,2,3,4,5,6,7,8,9,10,11,12,13,14,15);
    // independent chains fill each other's WMMA->WMMA hazard slots
    acc0 = __builtin_amdgcn_wmma_f32_16x16x32_f16(false, A0, false, B, (short)0, acc0, false, false);
    acc1 = __builtin_amdgcn_wmma_f32_16x16x32_f16(false, A1, false, B, (short)0, acc1, false, false);
  }
  // C layout: VGPR r -> row r (lanes 0-15) / 8+r (lanes 16-31); col = lane&15
  {
    const int rb = (lane >> 4) * 8;
    const int c  = w * 16 + (lane & 15);
#pragma unroll
    for (int r = 0; r < 8; ++r) mx_tile[(rb + r) * 132 + c]      = acc0[r];
#pragma unroll
    for (int r = 0; r < 8; ++r) mx_tile[(16 + rb + r) * 132 + c] = acc1[r];
  }
  __syncthreads();

  // ---- phase 2: per-row hyperbolic math; wave w owns rows 4w..4w+3 ----
  float pbv[4];
  float pb2l = 0.0f;
#pragma unroll
  for (int j = 0; j < 4; ++j) { pbv[j] = pb[lane + 32 * j]; pb2l += pbv[j] * pbv[j]; }
  const float pb2 = wave_sum32(pb2l);

  for (int lr = 4 * w; lr < 4 * w + 4; ++lr) {
    const int grow = row0 + lr;
    if (grow >= N) break;
    float mxv[4], xv[4], s_mx = 0.0f, s_x = 0.0f;
#pragma unroll
    for (int j = 0; j < 4; ++j) {
      mxv[j] = mx_tile[lr * 132 + lane + 32 * j];
      xv[j]  = x[(size_t)grow * D + lane + 32 * j];
      s_mx += mxv[j] * mxv[j];
      s_x  += xv[j] * xv[j];
    }
    const float mxn_raw = sqrtf(wave_sum32(s_mx));
    const float mxn = fmaxf(mxn_raw, MIN_NORM);
    const float xn  = fmaxf(sqrtf(wave_sum32(s_x)), MIN_NORM);
    // mobius_matvec scale (+ zero guard on ||mx|| <= 1e-10)
    const float sc = (mxn_raw <= 1e-10f) ? 0.0f : tanhf(mxn / xn * artanh_c(xn)) / mxn;
    float hv[4], s_h = 0.0f, s_xy = 0.0f;
#pragma unroll
    for (int j = 0; j < 4; ++j) { hv[j] = sc * mxv[j]; s_h += hv[j] * hv[j]; s_xy += hv[j] * pbv[j]; }
    const float x2 = wave_sum32(s_h);
    const float xy = wave_sum32(s_xy);
    // mobius_add(h, pb)
    const float ca  = 1.0f + 2.0f * xy + pb2;
    const float cb  = 1.0f - x2;
    const float den = fmaxf(1.0f + 2.0f * xy + x2 * pb2, MIN_NORM);
    float gv[4], s_g = 0.0f;
#pragma unroll
    for (int j = 0; j < 4; ++j) { gv[j] = (ca * hv[j] + cb * pbv[j]) / den; s_g += gv[j] * gv[j]; }
    // logmap0
    const float gn = fmaxf(sqrtf(wave_sum32(s_g)), MIN_NORM);
    const float ls = artanh_c(gn) / gn;
#pragma unroll
    for (int j = 0; j < 4; ++j) {
      const float o = ls * gv[j];
      h_t[(size_t)grow * D + lane + 32 * j] = o;
      agg[(size_t)grow * D + lane + 32 * j] = o;   // agg initialized to h_t (self term)
    }
  }
}

// ---- kernel 2: edge scatter  agg[dst] += h_t[src] * w_e  (L2-resident atomics) ----
__global__ void __launch_bounds__(256)
scatter_kernel(const float* __restrict__ h_t, const long long* __restrict__ eidx,
               const float* __restrict__ ew, float* __restrict__ agg, int E) {
  const int e = blockIdx.x * 8 + (threadIdx.x >> 5);  // one wave per edge
  if (e >= E) return;
  const int lane = threadIdx.x & 31;
  const int src = (int)eidx[e];
  const int dst = (int)eidx[(size_t)E + e];
  const float wgt = ew[e];
  const float4 v = *(const float4*)(h_t + (size_t)src * D + lane * 4);
  float* o = agg + (size_t)dst * D + lane * 4;
  atomicAdd(o + 0, v.x * wgt);
  atomicAdd(o + 1, v.y * wgt);
  atomicAdd(o + 2, v.z * wgt);
  atomicAdd(o + 3, v.w * wgt);
}

// ---- kernel 3: logmap0(expmap0(agg)) -> PReLU -> project(expmap0(.)) in place ----
__global__ void __launch_bounds__(256)
act_kernel(float* __restrict__ agg, const float* __restrict__ a_p, int N) {
  const int row = blockIdx.x * 8 + (threadIdx.x >> 5);
  if (row >= N) return;
  const int lane = threadIdx.x & 31;
  const float slope = a_p[0];
  float u[4], s = 0.0f;
#pragma unroll
  for (int j = 0; j < 4; ++j) { u[j] = agg[(size_t)row * D + lane + 32 * j]; s += u[j] * u[j]; }
  const float n1  = fmaxf(sqrtf(wave_sum32(s)), MIN_NORM);
  const float e1s = tanhf(n1) / n1;                       // expmap0
  float v[4], s2 = 0.0f;
#pragma unroll
  for (int j = 0; j < 4; ++j) { v[j] = e1s * u[j]; s2 += v[j] * v[j]; }
  const float n2  = fmaxf(sqrtf(wave_sum32(s2)), MIN_NORM);
  const float l1s = artanh_c(n2) / n2;                    // logmap0 (clipped)
  float tt[4], s3 = 0.0f;
#pragma unroll
  for (int j = 0; j < 4; ++j) {
    float z = l1s * v[j];
    z = (z >= 0.0f) ? z : slope * z;                      // PReLU
    tt[j] = z; s3 += z * z;
  }
  const float n3  = fmaxf(sqrtf(wave_sum32(s3)), MIN_NORM);
  const float e2s = tanhf(n3) / n3;                       // expmap0
  float o[4], s4 = 0.0f;
#pragma unroll
  for (int j = 0; j < 4; ++j) { o[j] = e2s * tt[j]; s4 += o[j] * o[j]; }
  const float n4 = fmaxf(sqrtf(wave_sum32(s4)), MIN_NORM);
  const float mn = 1.0f - BALL_EPS;
  const float ps = (n4 > mn) ? (mn / n4) : 1.0f;          // project
#pragma unroll
  for (int j = 0; j < 4; ++j) agg[(size_t)row * D + lane + 32 * j] = ps * o[j];
}

extern "C" void kernel_launch(void* const* d_in, const int* in_sizes, int n_in,
                              void* d_out, int out_size, void* d_ws, size_t ws_size,
                              hipStream_t stream) {
  const float*     x    = (const float*)d_in[0];
  const long long* eidx = (const long long*)d_in[1];   // int64 edge_index [2,E]
  const float*     ew   = (const float*)d_in[2];
  const float*     W    = (const float*)d_in[3];
  const float*     b    = (const float*)d_in[4];
  const float*     a    = (const float*)d_in[5];
  float* out = (float*)d_out;                          // doubles as agg buffer
  const int N = in_sizes[0] / D;
  const int E = in_sizes[2];

  // workspace layout: x16 (N*D f16) | W16 (D*D f16) | pb (D f32) | h_t (N*D f32)
  char* ws = (char*)d_ws;
  size_t off = 0;
  _Float16* x16 = (_Float16*)(ws + off); off += (size_t)N * D * 2; off = (off + 255) & ~(size_t)255;
  _Float16* W16 = (_Float16*)(ws + off); off += (size_t)D * D * 2; off = (off + 255) & ~(size_t)255;
  float*    pb  = (float*)(ws + off);    off += (size_t)D * 4;     off = (off + 255) & ~(size_t)255;
  float*    h_t = (float*)(ws + off);

  const int total4 = N * D / 4;
  cvt_x_f16<<<(total4 + 255) / 256, 256, 0, stream>>>(x, x16, total4);
  prep_wb<<<1, 128, 0, stream>>>(W, b, W16, pb);
  gemm_hyp_kernel<<<(N + 31) / 32, 256, 0, stream>>>(x, x16, W16, pb, h_t, out, N);
  scatter_kernel<<<(E + 7) / 8, 256, 0, stream>>>(h_t, eidx, ew, out, E);
  act_kernel<<<(N + 7) / 8, 256, 0, stream>>>(out, a, N);
}
